// HeteroAutoencoderMultiheadMean_919123001724
// MI455X (gfx1250) — matrix operand order repaired and checked by reading, not compiled, not run on hardware
//
#include <hip/hip_runtime.h>
#include <math.h>

// ---------------------------------------------------------------------------
// HeteroAutoencoderMultiheadMean for MI455X (gfx1250, wave32, WMMA).
// Dense GEMMs -> v_wmma_f32_16x16x32_f16 (f16 operands, f32 accum),
// double-buffered LDS staging, two-phase batched tile loads.
// Edge segment-softmax -> hardware float atomics in L2.
// Workspace requirement: ~1.92 GB (head-group chunking caps dec3 buffers).
// ---------------------------------------------------------------------------

#define H_HEADS 10
#define NA_N 30000
#define NT_N 60000
#define E_N  250000
#define ED_N 32
#define AF_N 336
#define TF_N 83

typedef __attribute__((ext_vector_type(16))) _Float16 v16h;
typedef __attribute__((ext_vector_type(8)))  _Float16 v8h;
typedef __attribute__((ext_vector_type(8)))  float    v8f;

static inline int ceil_div(int a, int b) { return (a + b - 1) / b; }

__device__ __forceinline__ void atomicMaxFloat(float* addr, float v) {
  // Exact for non-NaN floats: signed-int max for v>=0, unsigned min for v<0.
  if (v >= 0.f) atomicMax((int*)addr, __float_as_int(v));
  else          atomicMin((unsigned int*)addr, __float_as_uint(v));
}

// ---------------------------------------------------------------------------
// GEMM: Y[N,M] (+)= X[N,K] @ W[K,M](row stride ldW) + bias[M]
// Block = 256 threads = 8 wave32s. Grid = (ceil(N/16), ceil(M/128)).
// Wave w computes rows [r0,r0+16) x cols [n0+16w, n0+16w+16).
// Staging is branch-free, fully unrolled, and TWO-PHASE: all 18 global loads
// per tile are gathered into registers first (one batched loadcnt wait), then
// converted/packed into LDS. OOB rows/cols clamp to valid addresses (their
// products only reach discarded outputs; clamped data is finite so no NaN
// leakage); the summed k-tail is zeroed by select. Double-buffered LDS
// pipelines the next tile's loads behind the current WMMA.
// ---------------------------------------------------------------------------
__global__ __launch_bounds__(256)
void gemm_wmma_kernel(const float* __restrict__ X, int N, int K,
                      const float* __restrict__ W, int ldW,
                      const float* __restrict__ bias,
                      float* __restrict__ Y, int M, int acc)
{
  __shared__ __align__(16) _Float16 As[2][16 * 40];   // 16 rows x 32 k
  __shared__ __align__(16) _Float16 Bs[2][128 * 40];  // 128 cols x 32 k
  const int r0   = blockIdx.x * 16;
  const int n0   = blockIdx.y * 128;
  const int tid  = threadIdx.x;
  const int lane = tid & 31;
  const int wave = tid >> 5;
  const int l15  = lane & 15;
  const int hi16 = lane >> 4;
  const int ksteps = (K + 31) >> 5;

  // ---- k-step-invariant staging addresses ----
  // A panel: thread covers (m = tid>>5 + 8*it, k = lane), it = 0..1
  const int a_m0 = tid >> 5;
  int gr0 = r0 + a_m0;      gr0 = gr0 < N ? gr0 : N - 1;     // row clamp
  int gr1 = r0 + a_m0 + 8;  gr1 = gr1 < N ? gr1 : N - 1;
  const float* aRow0 = X + (size_t)gr0 * K;
  const float* aRow1 = X + (size_t)gr1 * K;
  // B panel: thread covers (n = tid&127, kp = (tid>>7) + 2*it), it = 0..7
  const int b_n = tid & 127;
  int b_gn = n0 + b_n;  b_gn = b_gn < M ? b_gn : M - 1;      // col clamp
  const float* wCol = W + b_gn;
  const int b_kp0 = tid >> 7;

  auto load_tile = [&](int ks, int buf) {
    const int kt = ks << 5;
    const int gk  = kt + lane;
    const int gkc = gk < K ? gk : K - 1;                     // addr clamp
    const bool kin = gk < K;

    // ---- phase 1: gather everything (loads issue back-to-back) ----
    float a0 = aRow0[gkc];
    float a1 = aRow1[gkc];
    float w0[8], w1[8];
#pragma unroll
    for (int it = 0; it < 8; ++it) {
      int kp = b_kp0 + 2 * it;
      int k0 = kt + 2 * kp, k1 = k0 + 1;
      int k0c = k0 < K ? k0 : K - 1;
      int k1c = k1 < K ? k1 : K - 1;
      w0[it] = wCol[(size_t)k0c * ldW];
      w1[it] = wCol[(size_t)k1c * ldW];
    }

    // ---- phase 2: convert + store to LDS ----
    As[buf][a_m0 * 40 + lane]       = (_Float16)(kin ? a0 : 0.f);
    As[buf][(a_m0 + 8) * 40 + lane] = (_Float16)(kin ? a1 : 0.f);
#pragma unroll
    for (int it = 0; it < 8; ++it) {
      int kp = b_kp0 + 2 * it;
      int k0 = kt + 2 * kp, k1 = k0 + 1;
      union { _Float16 h[2]; unsigned u; } pk;
      pk.h[0] = (_Float16)(k0 < K ? w0[it] : 0.f);
      pk.h[1] = (_Float16)(k1 < K ? w1[it] : 0.f);
      *(unsigned*)&Bs[buf][b_n * 40 + 2 * kp] = pk.u;
    }
  };

  v8f accv = {};
  load_tile(0, 0);
  __syncthreads();

  for (int ks = 0; ks < ksteps; ++ks) {
    const int cur = ks & 1;
    // Issue this step's operand LDS reads first (LDS is in-order per wave,
    // so they complete ahead of the next tile's ds_stores).
    // A f16 16x32: lanes 0-15 row=l, halves0-7->K 0-7, halves8-15->K 16-23;
    //              lanes16-31 row=l-16, K 8-15 / 24-31.
    union { v16h v; v8h h[2]; } a, b;
    const _Float16* arow = &As[cur][l15 * 40];
    a.h[0] = *(const v8h*)(arow + 8 * hi16);
    a.h[1] = *(const v8h*)(arow + 16 + 8 * hi16);
    // B f16 32x16: lane N=l&15; lanes0-15 K 0-7/16-23, lanes16-31 K 8-15/24-31.
    const _Float16* brow = &Bs[cur][(wave * 16 + l15) * 40 + 8 * hi16];
    b.h[0] = *(const v8h*)(brow);
    b.h[1] = *(const v8h*)(brow + 16);

    if (ks + 1 < ksteps) {
      load_tile(ks + 1, cur ^ 1);
      // prefetch one more tile ahead (global_prefetch on CDNA5)
      if (ks + 2 < ksteps) {
        int gk2 = ((ks + 2) << 5) + (tid >> 3);
        if (gk2 < K) __builtin_prefetch(&wCol[(size_t)gk2 * ldW], 0, 1);
      }
    }

    accv = __builtin_amdgcn_wmma_f32_16x16x32_f16(false, a.v, false, b.v,
                                                  (short)0, accv, false, false);
    __syncthreads();
  }

  // C/D layout: vgpr v, lanes0-15 -> (M=v, N=lane); lanes16-31 -> (M=v+8).
  const int col = n0 + wave * 16 + l15;
  if (col < M) {
    const float bv = bias ? bias[col] : 0.f;
#pragma unroll
    for (int v = 0; v < 8; ++v) {
      const int row = r0 + v + 8 * hi16;
      if (row < N) {
        size_t idx = (size_t)row * M + col;
        float r = accv[v] + bv;
        Y[idx] = acc ? (Y[idx] + r) : r;
      }
    }
  }
}

// ---------------------------------------------------------------------------
__global__ void fill_kernel(float* __restrict__ p, size_t n, float v) {
  size_t i = (size_t)blockIdx.x * blockDim.x + threadIdx.x;
  if (i < n) p[i] = v;
}

// logit[e,h] = scale * <Q[dst], K[src] (+ Ee[e])>, plus running segment max.
__global__ void edge_logits_kernel(const float* __restrict__ Q, const float* __restrict__ Kt,
                                   const float* __restrict__ Ee,
                                   const int* __restrict__ src, const int* __restrict__ dst,
                                   int ne, int Hg, int C, int ld, float scale,
                                   float* __restrict__ logits, float* __restrict__ mbuf)
{
  int i = blockIdx.x * blockDim.x + threadIdx.x;
  if (i >= ne * Hg) return;
  int e = i / Hg;
  int h = i - e * Hg;
  int s = src[e], d = dst[e];
  const float* q = Q  + (size_t)d * ld + h * C;
  const float* k = Kt + (size_t)s * ld + h * C;
  float accd = 0.f;
  if (Ee) {
    const float* ee = Ee + (size_t)e * ld + h * C;
#pragma unroll 4
    for (int c = 0; c < C; ++c) accd += q[c] * (k[c] + ee[c]);
  } else {
#pragma unroll 4
    for (int c = 0; c < C; ++c) accd += q[c] * k[c];
  }
  float lg = accd * scale;
  logits[i] = lg;
  atomicMaxFloat(&mbuf[(size_t)d * Hg + h], lg);
}

__global__ void edge_exden_kernel(const float* __restrict__ logits, const float* __restrict__ mbuf,
                                  const int* __restrict__ dst, int ne, int Hg,
                                  float* __restrict__ ex, float* __restrict__ den)
{
  int i = blockIdx.x * blockDim.x + threadIdx.x;
  if (i >= ne * Hg) return;
  int e = i / Hg;
  int h = i - e * Hg;
  int d = dst[e];
  float x = __expf(logits[i] - mbuf[(size_t)d * Hg + h]);
  ex[i] = x;
  atomicAdd(&den[(size_t)d * Hg + h], x);
}

// agg[dst, h*C+c] += alpha * (V[src] (+ Ee[e]))
__global__ void edge_agg_kernel(const float* __restrict__ ex, const float* __restrict__ den,
                                const float* __restrict__ V, const float* __restrict__ Ee,
                                const int* __restrict__ src, const int* __restrict__ dst,
                                int ne, int Hg, int C, int ld,
                                float* __restrict__ agg, int ldAgg)
{
  int i = blockIdx.x * blockDim.x + threadIdx.x;
  if (i >= ne * Hg) return;
  int e = i / Hg;
  int h = i - e * Hg;
  int s = src[e], d = dst[e];
  float alpha = ex[i] / (den[(size_t)d * Hg + h] + 1e-16f);
  const float* v = V + (size_t)s * ld + h * C;
  float* ag = agg + (size_t)d * ldAgg + h * C;
  if (Ee) {
    const float* ee = Ee + (size_t)e * ld + h * C;
    for (int c = 0; c < C; ++c) atomicAdd(&ag[c], alpha * (v[c] + ee[c]));
  } else {
    for (int c = 0; c < C; ++c) atomicAdd(&ag[c], alpha * v[c]);
  }
}

// out = maybe_relu( concat? agg + S : mean_heads(agg) + S )
__global__ void finalize_kernel(const float* __restrict__ agg, const float* __restrict__ S,
                                float* __restrict__ out, int n, int d, int C,
                                int concat, int relu)
{
  int tot = concat ? n * d : n * C;
  int i = blockIdx.x * blockDim.x + threadIdx.x;
  if (i >= tot) return;
  float r;
  if (concat) {
    r = agg[i] + S[i];
  } else {
    int row = i / C, c = i - row * C;
    const float* ar = agg + (size_t)row * d + c;
    float s = 0.f;
#pragma unroll
    for (int h = 0; h < H_HEADS; ++h) s += ar[h * C];
    r = s * (1.f / H_HEADS) + S[i];
  }
  out[i] = relu ? fmaxf(r, 0.f) : r;
}

__global__ void colsum_kernel(const float* __restrict__ X, int N, int D, float* __restrict__ cs)
{
  int c = blockIdx.x * blockDim.x + threadIdx.x;
  if (c >= D) return;
  int r0 = blockIdx.y * 256;
  int r1 = (r0 + 256 < N) ? (r0 + 256) : N;
  float s = 0.f;
  for (int r = r0; r < r1; ++r) s += X[(size_t)r * D + c];
  atomicAdd(&cs[c], s);
}

__global__ void bcast_relu_kernel(const float* __restrict__ cs, int N, int D, float inv,
                                  float* __restrict__ out)
{
  int i = blockIdx.x * blockDim.x + threadIdx.x;
  if (i >= N * D) return;
  float v = cs[i % D] * inv;
  out[i] = v > 0.f ? v : 0.f;
}

// ---------------------------------------------------------------------------
// Host orchestration
// ---------------------------------------------------------------------------
struct RelP {
  const float *We, *Wk, *Wq, *Ws, *Wv, *be, *bk, *bq, *bs, *bv;
  int in_src, in_dst, C, d, so;
  bool edge;
};
struct LayerP { RelP ends, receives, sends, starts; bool concat; };

extern "C" void kernel_launch(void* const* d_in, const int* in_sizes, int n_in,
                              void* d_out, int out_size, void* d_ws, size_t ws_size,
                              hipStream_t stream) {
  (void)in_sizes; (void)n_in; (void)out_size; (void)ws_size;

  const float* P     = (const float*)d_in[0];
  const float* xaddr = (const float*)d_in[1];
  const float* xtran = (const float*)d_in[2];
  const int* ss = (const int*)d_in[3];  const int* sd = (const int*)d_in[4];
  const int* rs = (const int*)d_in[5];  const int* rd = (const int*)d_in[6];
  const int* ts = (const int*)d_in[7];  const int* td = (const int*)d_in[8];
  const int* es = (const int*)d_in[9];  const int* ed = (const int*)d_in[10];
  const float* eas = (const float*)d_in[11];
  const float* eae = (const float*)d_in[12];

  // ---- parse flattened params (JAX tree_flatten: keys sorted, 'W*' < 'b*')
  size_t cur = 0;
  auto take = [&](size_t n) { const float* p = P + cur; cur += n; return p; };
  auto parse_rel = [&](int in_src, int in_dst, int C, bool concat, bool edge) {
    RelP r{};
    r.in_src = in_src; r.in_dst = in_dst; r.C = C;
    r.d = H_HEADS * C; r.so = concat ? r.d : C; r.edge = edge;
    if (edge) r.We = take((size_t)ED_N * r.d);
    r.Wk = take((size_t)in_src * r.d);
    r.Wq = take((size_t)in_dst * r.d);
    r.Ws = take((size_t)in_dst * r.so);
    r.Wv = take((size_t)in_src * r.d);
    if (edge) r.be = take((size_t)r.d);
    r.bk = take((size_t)r.d); r.bq = take((size_t)r.d);
    r.bs = take((size_t)r.so); r.bv = take((size_t)r.d);
    return r;
  };
  auto parse_layer = [&](int in_a, int in_t, int c_t, int c_a, bool concat) {
    LayerP L; L.concat = concat;
    L.ends     = parse_rel(in_t, in_a, c_a, concat, true);
    L.receives = parse_rel(in_t, in_a, c_a, concat, false);
    L.sends    = parse_rel(in_a, in_t, c_t, concat, false);
    L.starts   = parse_rel(in_a, in_t, c_t, concat, true);
    return L;
  };
  LayerP dec1 = parse_layer(120, 120, 12, 12, true);
  LayerP dec2 = parse_layer(120, 120, 12, 12, true);
  LayerP dec3 = parse_layer(120, 120, TF_N, AF_N, false);
  LayerP enc1 = parse_layer(AF_N, TF_N, 12, 12, true);
  LayerP enc2 = parse_layer(120, 120, 12, 12, true);
  LayerP enc3 = parse_layer(120, 120, 12, 12, true);

  // ---- workspace layout (floats), ~479M floats total (~1.92 GB)
  float* wsf = (float*)d_ws;
  size_t off = 0;
  auto wsalloc = [&](size_t n) { float* p = wsf + off; off += n; return p; };
  float* Qg   = wsalloc(50000000ull);   // <= 60000*830
  float* Kg   = wsalloc(41000000ull);   // <= 60000*672
  float* Vg   = wsalloc(41000000ull);
  float* Eeg  = wsalloc(208000000ull);  // <= 250000*830
  float* Sbuf = wsalloc(10500000ull);   // <= 30000*336
  float* aggb = wsalloc(101000000ull);  // <= 30000*3360
  float* lgb  = wsalloc(2500000ull);    // E*Hg
  float* exb  = wsalloc(2500000ull);
  float* mbuf = wsalloc(600000ull);     // n_dst*Hg
  float* dbuf = wsalloc(600000ull);
  float* aA = wsalloc((size_t)NA_N * 120);
  float* aB = wsalloc((size_t)NA_N * 120);
  float* tA = wsalloc((size_t)NT_N * 120);
  float* tB = wsalloc((size_t)NT_N * 120);
  float* csum = wsalloc(128);

  // ---- launch helpers
  auto gemm = [&](const float* X, int N, int K, const float* W, int ldW,
                  const float* b, float* Y, int M, int acc) {
    dim3 g(ceil_div(N, 16), ceil_div(M, 128));
    gemm_wmma_kernel<<<g, 256, 0, stream>>>(X, N, K, W, ldW, b, Y, M, acc);
  };
  auto fill = [&](float* p, size_t n, float v) {
    int blocks = (int)((n + 255) / 256);
    fill_kernel<<<blocks, 256, 0, stream>>>(p, n, v);
  };

  auto do_rel = [&](const RelP& r, const float* xs, int fs, const float* xd, int fd,
                    int n_src, int n_dst, const int* srci, const int* dsti,
                    const float* ea) {
    // head-group size so that the per-edge embedding slice stays <= 1 GB
    int Hg = H_HEADS;
    while (Hg > 1 && (size_t)E_N * 4 * r.C * Hg > (1ull << 30)) Hg /= 2;
    const int nGroups = H_HEADS / Hg;
    const float scale = 1.f / sqrtf((float)r.C);
    for (int g = 0; g < nGroups; ++g) {
      const int co = g * Hg * r.C;       // column offset into W*/b*
      const int dg = Hg * r.C;
      gemm(xd, n_dst, fd, r.Wq + co, r.d, r.bq + co, Qg, dg, 0);
      gemm(xs, n_src, fs, r.Wk + co, r.d, r.bk + co, Kg, dg, 0);
      gemm(xs, n_src, fs, r.Wv + co, r.d, r.bv + co, Vg, dg, 0);
      const float* EePtr = nullptr;
      if (r.edge) {
        gemm(ea, E_N, ED_N, r.We + co, r.d, r.be + co, Eeg, dg, 0);
        EePtr = Eeg;
      }
      fill(mbuf, (size_t)n_dst * Hg, -INFINITY);
      fill(dbuf, (size_t)n_dst * Hg, 0.f);
      const int tot = E_N * Hg;
      edge_logits_kernel<<<ceil_div(tot, 256), 256, 0, stream>>>(
          Qg, Kg, EePtr, srci, dsti, E_N, Hg, r.C, dg, scale, lgb, mbuf);
      edge_exden_kernel<<<ceil_div(tot, 256), 256, 0, stream>>>(
          lgb, mbuf, dsti, E_N, Hg, exb, dbuf);
      edge_agg_kernel<<<ceil_div(tot, 256), 256, 0, stream>>>(
          exb, dbuf, Vg, EePtr, srci, dsti, E_N, Hg, r.C, dg, aggb + co, r.d);
    }
  };

  auto do_dst = [&](const RelP& r1, const int* s1, const int* d1, const float* ea1,
                    const RelP& r2, const int* s2, const int* d2, const float* ea2,
                    const float* xs, int fs, int n_src,
                    const float* xd, int fd, int n_dst,
                    float* out, bool concat, bool relu) {
    const int dfull = r1.d, so = r1.so;
    // skip (root) terms of both relations, summed
    gemm(xd, n_dst, fd, r1.Ws, so, r1.bs, Sbuf, so, 0);
    gemm(xd, n_dst, fd, r2.Ws, so, r2.bs, Sbuf, so, 1);
    // attention aggregates of both relations, summed (HeteroConv 'sum')
    fill(aggb, (size_t)n_dst * dfull, 0.f);
    do_rel(r1, xs, fs, xd, fd, n_src, n_dst, s1, d1, ea1);
    do_rel(r2, xs, fs, xd, fd, n_src, n_dst, s2, d2, ea2);
    const int tot = concat ? n_dst * dfull : n_dst * r1.C;
    finalize_kernel<<<ceil_div(tot, 256), 256, 0, stream>>>(
        aggb, Sbuf, out, n_dst, dfull, r1.C, concat ? 1 : 0, relu ? 1 : 0);
  };

  auto het = [&](const LayerP& L, const float* xa_, int fa_, const float* xt_, int ft_,
                 float* aout, float* tout, bool relu) {
    // t-dst: sends (a->t) + starts (a->t, edge attrs)
    do_dst(L.sends, ss, sd, nullptr, L.starts, ts, td, eas,
           xa_, fa_, NA_N, xt_, ft_, NT_N, tout, L.concat, relu);
    // a-dst: receives (t->a) + ends (t->a, edge attrs)
    do_dst(L.receives, rs, rd, nullptr, L.ends, es, ed, eae,
           xt_, ft_, NT_N, xa_, fa_, NA_N, aout, L.concat, relu);
  };

  // ---- forward
  het(enc1, xaddr, AF_N, xtran, TF_N, aA, tA, true);
  het(enc2, aA, 120, tA, 120, aB, tB, true);
  het(enc3, aB, 120, tB, 120, aA, tA, false);

  // mean-pool over nodes, relu, broadcast back to node counts
  fill(csum, 128, 0.f);
  colsum_kernel<<<dim3(1, ceil_div(NA_N, 256)), 128, 0, stream>>>(aA, NA_N, 120, csum);
  bcast_relu_kernel<<<ceil_div(NA_N * 120, 256), 256, 0, stream>>>(
      csum, NA_N, 120, 1.f / NA_N, aB);
  fill(csum, 128, 0.f);
  colsum_kernel<<<dim3(1, ceil_div(NT_N, 256)), 128, 0, stream>>>(tA, NT_N, 120, csum);
  bcast_relu_kernel<<<ceil_div(NT_N * 120, 256), 256, 0, stream>>>(
      csum, NT_N, 120, 1.f / NT_N, tB);

  het(dec1, aB, 120, tB, 120, aA, tA, true);
  het(dec2, aA, 120, tA, 120, aB, tB, true);

  float* outA = (float*)d_out;
  float* outT = outA + (size_t)NA_N * AF_N;
  het(dec3, aB, 120, tB, 120, outA, outT, false);
}
